// Model_1_26972394619434
// MI455X (gfx1250) — compile-verified, matrix operand order checked by reading, not run on hardware
//
#include <hip/hip_runtime.h>

typedef __attribute__((ext_vector_type(16))) _Float16 v16h;
typedef __attribute__((ext_vector_type(8)))  float    v8f;
typedef __attribute__((ext_vector_type(4)))  float    v4f;

#define T_STEPS   100
#define KDIM      784
#define NCHUNK    25          // ceil(784/32) K-chunks of 32 for f16 WMMA
#define WAVES     8
#define THREADS   (WAVES * 32)
#define HALF_N    6422528u    // (16384*784)/2 : JAX threefry counter pairing offset

// JAX threefry-2x32 (20 rounds, standard rotation schedule + key injection)
__device__ __forceinline__ uint2 threefry2x32(unsigned k0, unsigned k1,
                                              unsigned x0, unsigned x1) {
  unsigned ks2 = k0 ^ k1 ^ 0x1BD11BDAu;
  x0 += k0; x1 += k1;
#define TF_R(r) { x0 += x1; x1 = (x1 << (r)) | (x1 >> (32 - (r))); x1 ^= x0; }
  TF_R(13) TF_R(15) TF_R(26) TF_R(6)   x0 += k1;  x1 += ks2 + 1u;
  TF_R(17) TF_R(29) TF_R(16) TF_R(24)  x0 += ks2; x1 += k0  + 2u;
  TF_R(13) TF_R(15) TF_R(26) TF_R(6)   x0 += k0;  x1 += k1  + 3u;
  TF_R(17) TF_R(29) TF_R(16) TF_R(24)  x0 += k1;  x1 += ks2 + 4u;
  TF_R(13) TF_R(15) TF_R(26) TF_R(6)   x0 += ks2; x1 += k0  + 5u;
#undef TF_R
  uint2 r; r.x = x0; r.y = x1; return r;
}

// JAX uniform[0,1): bits>>9 | 0x3f800000, bitcast, minus 1
__device__ __forceinline__ float u01(unsigned bits) {
  return __uint_as_float((bits >> 9) | 0x3F800000u) - 1.0f;
}

// Full K-chunk (all 32 K slots provably in-bounds): 16 threefry calls ->
// spikes for rows b0 and b0+8192 simultaneously, then two f16 WMMAs.
#define DO_CHUNK(KB, BM)                                                       \
  do {                                                                         \
    const v4f* pa = (const v4f*)(xf + f0 + (unsigned)(KB));                    \
    const v4f* pb = (const v4f*)(xf + f0 + HALF_N + (unsigned)(KB));           \
    const v4f qa0 = pa[0], qa1 = pa[1], qa2 = pa[4], qa3 = pa[5];              \
    const v4f qb0 = pb[0], qb1 = pb[1], qb2 = pb[4], qb3 = pb[5];              \
    v16h a0, a1;                                                               \
    _Pragma("unroll")                                                          \
    for (int i = 0; i < 16; ++i) {                                             \
      const unsigned j = (unsigned)(KB) + (unsigned)((i & 7) + ((i >> 3) << 4));\
      uint2 r = threefry2x32(key0, key1, f0 + j, f0 + j + HALF_N);             \
      const float xa = (i < 4 ? qa0[i & 3] : i < 8 ? qa1[i & 3]                \
                        : i < 12 ? qa2[i & 3] : qa3[i & 3]);                   \
      const float xb = (i < 4 ? qb0[i & 3] : i < 8 ? qb1[i & 3]                \
                        : i < 12 ? qb2[i & 3] : qb3[i & 3]);                   \
      a0[i] = (_Float16)((u01(r.x) < xa) ? 1.0f : 0.0f);                       \
      a1[i] = (_Float16)((u01(r.y) < xb) ? 1.0f : 0.0f);                       \
    }                                                                          \
    p0 = __builtin_amdgcn_wmma_f32_16x16x32_f16(false, a0, false, (BM),        \
                                                (short)0, p0, false, false);   \
    p1 = __builtin_amdgcn_wmma_f32_16x16x32_f16(false, a1, false, (BM),        \
                                                (short)0, p1, false, false);   \
  } while (0)

// Tail chunk 24: K slots i>=8 map to j>=784 for ALL lanes -> hardwired zero,
// only 8 threefry calls. Runs on wave 0 only (balances 25 chunks over 8 waves).
#define DO_HALF24(KB, BM)                                                      \
  do {                                                                         \
    const v4f* pa = (const v4f*)(xf + f0 + (unsigned)(KB));                    \
    const v4f* pb = (const v4f*)(xf + f0 + HALF_N + (unsigned)(KB));           \
    const v4f qa0 = pa[0], qa1 = pa[1];                                        \
    const v4f qb0 = pb[0], qb1 = pb[1];                                        \
    v16h a0, a1;                                                               \
    _Pragma("unroll")                                                          \
    for (int i = 0; i < 16; ++i) {                                             \
      if (i < 8) {                                                             \
        const unsigned j = (unsigned)(KB) + (unsigned)i;                       \
        uint2 r = threefry2x32(key0, key1, f0 + j, f0 + j + HALF_N);           \
        const float xa = (i < 4 ? qa0[i & 3] : qa1[i & 3]);                    \
        const float xb = (i < 4 ? qb0[i & 3] : qb1[i & 3]);                    \
        a0[i] = (_Float16)((u01(r.x) < xa) ? 1.0f : 0.0f);                     \
        a1[i] = (_Float16)((u01(r.y) < xb) ? 1.0f : 0.0f);                     \
      } else {                                                                 \
        a0[i] = (_Float16)0.0f;                                                \
        a1[i] = (_Float16)0.0f;                                                \
      }                                                                        \
    }                                                                          \
    p0 = __builtin_amdgcn_wmma_f32_16x16x32_f16(false, a0, false, (BM),        \
                                                (short)0, p0, false, false);   \
    p1 = __builtin_amdgcn_wmma_f32_16x16x32_f16(false, a1, false, (BM),        \
                                                (short)0, p1, false, false);   \
  } while (0)

__global__ __launch_bounds__(THREADS) void snn_lif_poisson_kernel(
    const float* __restrict__ xf,   // [16384, 784] pixel intensities
    const float* __restrict__ W,    // [10, 784]
    float* __restrict__ out)        // [16384, 10] mean spike rate
{
  // W pre-swizzled into WMMA B-fragment order, f16, zero-padded (K->800, N->16)
  __shared__ __align__(32) unsigned short bfrag[NCHUNK][32][16];
  // double-buffered per-wave partial currents (16x16 f32 C-tiles x 2 row groups)
  __shared__ float part[2][WAVES][2][256];

  const int tid  = threadIdx.x;
  const int wave = __builtin_amdgcn_readfirstlane(tid) >> 5;  // SGPR-resident
  const int lane = tid & 31;

  // ---- Stage W as B fragments (once). B 32x16 f16 layout mirrors A:
  // lane l: N = l%16; half i holds K = (i&7) + (i>=8 ? 16 : 0) + (l>=16 ? 8 : 0)
  for (int idx = tid; idx < NCHUNK * 32; idx += THREADS) {
    const int c  = idx >> 5;
    const int l  = idx & 31;
    const int n  = l & 15;
    const int kb = c * 32 + ((l >> 4) << 3);
#pragma unroll
    for (int i = 0; i < 16; ++i) {
      const int j = kb + (i & 7) + ((i >> 3) << 4);
      float w = (n < 10 && j < KDIM) ? W[n * KDIM + j] : 0.0f;
      _Float16 h = (_Float16)w;
      bfrag[c][l][i] = __builtin_bit_cast(unsigned short, h);
    }
  }
  __syncthreads();

  const unsigned b0    = (unsigned)(blockIdx.x * 16 + (lane & 15)); // [0,8192)
  const unsigned f0    = b0 * (unsigned)KDIM;                       // flat base
  const int      laneK = (lane >> 4) << 3;                          // +0 / +8

  // Static chunk assignment: wave w owns chunks {w, w+8, w+16}; wave 0 also
  // owns the half-valid tail chunk 24. All B fragments live in VGPRs.
  const int kb0 = wave * 32 + laneK;
  const int kb1 = kb0 + 256;
  const int kb2 = kb0 + 512;
  const int kb3 = 768 + laneK;

  const v16h bm0 = *(const v16h*)&bfrag[wave][lane][0];
  const v16h bm1 = *(const v16h*)&bfrag[wave + 8][lane][0];
  const v16h bm2 = *(const v16h*)&bfrag[wave + 16][lane][0];
  const v16h bm3 = *(const v16h*)&bfrag[24][lane][0];

  v8f v0 = {}, v1 = {}, acc0 = {}, acc1 = {};

  for (int t = 0; t < T_STEPS; ++t) {
    // keys[t] from split(key(42), 100) = threefry(master, iota(200)) pairs
    unsigned key0, key1;
    {
      const unsigned i0 = 2u * (unsigned)t, i1 = i0 + 1u;
      if (i0 < 100u) {
        key0 = threefry2x32(0u, 42u, i0, i0 + 100u).x;
        key1 = threefry2x32(0u, 42u, i1, i1 + 100u).x;
      } else {
        key0 = threefry2x32(0u, 42u, i0 - 100u, i0).y;
        key1 = threefry2x32(0u, 42u, i1 - 100u, i1).y;
      }
    }

    // ---- Poisson-encode + GEMM partial currents for this wave's chunks ----
    v8f p0 = {}, p1 = {};
    DO_CHUNK(kb0, bm0);
    DO_CHUNK(kb1, bm1);
    DO_CHUNK(kb2, bm2);
    if (wave == 0) {          // scalar branch: EXEC stays all-ones inside
      DO_HALF24(kb3, bm3);
    }

    // ---- cross-wave K reduction (double-buffered, 1 barrier per step) ----
    const int buf = t & 1;
#pragma unroll
    for (int r = 0; r < 8; ++r) {
      part[buf][wave][0][r * 32 + lane] = p0[r];
      part[buf][wave][1][r * 32 + lane] = p1[r];
    }
    __syncthreads();

    // ---- replicated LIF update (all waves keep identical v/acc state) ----
#pragma unroll
    for (int r = 0; r < 8; ++r) {
      float I0 = 0.0f, I1 = 0.0f;
#pragma unroll
      for (int w = 0; w < WAVES; ++w) {
        I0 += part[buf][w][0][r * 32 + lane];
        I1 += part[buf][w][1][r * 32 + lane];
      }
      // v += (I - v)/tau, tau=2 ; spike at v>=1 ; hard reset to 0
      float nv0 = v0[r] + (I0 - v0[r]) * 0.5f;
      float s0  = (nv0 >= 1.0f) ? 1.0f : 0.0f;
      acc0[r] += s0;  v0[r] = nv0 * (1.0f - s0);

      float nv1 = v1[r] + (I1 - v1[r]) * 0.5f;
      float s1  = (nv1 >= 1.0f) ? 1.0f : 0.0f;
      acc1[r] += s1;  v1[r] = nv1 * (1.0f - s1);
    }
  }

  // ---- epilogue: C-tile layout lane l, vgpr r -> (M = r + 8*(l>=16), N = l%16)
  if (wave == 0) {
    const int n = lane & 15;
    if (n < 10) {
#pragma unroll
      for (int r = 0; r < 8; ++r) {
        const int row0 = blockIdx.x * 16 + r + ((lane >> 4) << 3);
        out[row0 * 10 + n]          = acc0[r] * (1.0f / T_STEPS);
        out[(row0 + 8192) * 10 + n] = acc1[r] * (1.0f / T_STEPS);
      }
    }
  }
}

extern "C" void kernel_launch(void* const* d_in, const int* in_sizes, int n_in,
                              void* d_out, int out_size, void* d_ws, size_t ws_size,
                              hipStream_t stream) {
  (void)in_sizes; (void)n_in; (void)out_size; (void)d_ws; (void)ws_size;
  const float* xf = (const float*)d_in[0];   // [16384,1,28,28] -> flat [16384,784]
  const float* W  = (const float*)d_in[1];   // [10,784]
  float* out      = (float*)d_out;           // [16384,10]
  // 512 blocks: one 16-row tile of b in [0,8192) paired with b+8192 per block
  snn_lif_poisson_kernel<<<512, THREADS, 0, stream>>>(xf, W, out);
}
